// Decoder_17463337026012
// MI455X (gfx1250) — matrix operand order repaired and checked by reading, not compiled, not run on hardware
//
#include <hip/hip_runtime.h>
#include <stdint.h>

typedef __bf16 bf16;
typedef __attribute__((ext_vector_type(16))) __bf16 v16bf;
typedef __attribute__((ext_vector_type(8)))  float  v8f;

#define NBLK 64
#define NTHR 256

__device__ __forceinline__ float sigm(float x) {
  return 1.0f / (1.0f + __expf(-x));
}
__device__ __forceinline__ float tanh_f(float x) {
  float e = __expf(2.0f * x);
  return (e - 1.0f) / (e + 1.0f);
}

// Piecewise source for the A matrix (concatenated activations).
struct ASrc {
  const float* p0; const float* p1; const float* p2;
  int e0, e1;        // K boundaries (multiples of 512)
  int s0, s1, s2;    // row strides
};

// Grid-wide barrier: release-add + acquire-spin (agent scope) with a
// monotonically increasing counter, reset to 0 by hipMemsetAsync per launch.
__device__ __forceinline__ void grid_barrier(unsigned* bar, unsigned target) {
  __syncthreads();
  if (threadIdx.x == 0) {
    __hip_atomic_fetch_add(bar, 1u, __ATOMIC_RELEASE, __HIP_MEMORY_SCOPE_AGENT);
    while (__hip_atomic_load(bar, __ATOMIC_ACQUIRE, __HIP_MEMORY_SCOPE_AGENT) < target)
      __builtin_amdgcn_s_sleep(1);
  }
  __syncthreads();
}

// Pack a 64-row x 256-K chunk of A into LDS in the exact WMMA A-operand
// layout (16-bit A 16x32: lane<16 -> K {0..7,16..23}, lane>=16 -> +8).
// A 32-wide k-tile never straddles a segment boundary (segments are
// multiples of 512), so resolve the source pointer once per fragment and
// read 4 aligned float4s (global_load_b128).
__device__ __forceinline__ void packA(v16bf* Abuf, const ASrc& a, int kbase) {
  const int tid = threadIdx.x;
  #pragma unroll
  for (int q = 0; q < 4; ++q) {
    int fi  = q * NTHR + tid;
    int pmt = fi >> 8;
    int pkt = (fi >> 5) & 7;
    int pl  = fi & 31;
    int m   = pmt * 16 + (pl & 15);
    int ko  = (pl < 16) ? 0 : 8;
    int K0  = kbase + pkt * 32;
    const float* src;
    if (K0 < a.e0)      src = a.p0 + m * a.s0 + K0;
    else if (K0 < a.e1) src = a.p1 + m * a.s1 + (K0 - a.e0);
    else                src = a.p2 + m * a.s2 + (K0 - a.e1);
    src += ko;                       // 32B-aligned (K0 mult of 32, ko in {0,8})
    float4 fa = *(const float4*)(src);
    float4 fb = *(const float4*)(src + 4);
    float4 fc = *(const float4*)(src + 16);
    float4 fd = *(const float4*)(src + 20);
    v16bf frag;
    frag[0]  = (bf16)fa.x; frag[1]  = (bf16)fa.y;
    frag[2]  = (bf16)fa.z; frag[3]  = (bf16)fa.w;
    frag[4]  = (bf16)fb.x; frag[5]  = (bf16)fb.y;
    frag[6]  = (bf16)fb.z; frag[7]  = (bf16)fb.w;
    frag[8]  = (bf16)fc.x; frag[9]  = (bf16)fc.y;
    frag[10] = (bf16)fc.z; frag[11] = (bf16)fc.w;
    frag[12] = (bf16)fd.x; frag[13] = (bf16)fd.y;
    frag[14] = (bf16)fd.z; frag[15] = (bf16)fd.w;
    Abuf[fi] = frag;
  }
}

// One LSTM layer for this block: GEMM over the block's four 16-wide gate
// strips (i,j,f,o at N = blk*16 + [0,16) within each 1024 quadrant) fused
// with the peephole cell update for U-columns [blk*16, blk*16+16).
__device__ void gemm_cell(v16bf* Abuf, float (*gates)[64][16],
                          const ASrc& asrc, const v16bf* Bp,
                          const float* bias, const float* pi, const float* pf,
                          const float* po, float* c, float* hw, int blk)
{
  const int tid  = threadIdx.x;
  const int lane = tid & 31;
  const int wv   = tid >> 5;
  const int mt   = wv & 3;        // M tile (batch rows 16*mt..)
  const int g0   = (wv >> 2) * 2; // gate pair: waves 0-3 -> gates 0,1; 4-7 -> 2,3
  const size_t strip0 = (size_t)((g0 + 0) * 64 + blk) * 64;
  const size_t strip1 = (size_t)((g0 + 1) * 64 + blk) * 64;
  v8f acc0 = {};
  v8f acc1 = {};
  for (int ch = 0; ch < 8; ++ch) {          // K chunks of 256 (K = 2048)
    __syncthreads();
    // Warm the WGP cache with this chunk's weight fragments while packing A.
    #pragma unroll
    for (int pk = 0; pk < 8; ++pk) {
      __builtin_prefetch(&Bp[(strip0 + ch * 8 + pk) * 32 + lane], 0, 1);
      __builtin_prefetch(&Bp[(strip1 + ch * 8 + pk) * 32 + lane], 0, 1);
    }
    packA(Abuf, asrc, ch * 256);
    __syncthreads();
    #pragma unroll 1                        // keep loads next to WMMAs: no spills
    for (int ktl = 0; ktl < 8; ++ktl) {
      int kt = ch * 8 + ktl;
      v16bf a  = Abuf[(mt * 8 + ktl) * 32 + lane];
      v16bf b0 = Bp[(strip0 + kt) * 32 + lane];
      v16bf b1 = Bp[(strip1 + kt) * 32 + lane];
      acc0 = __builtin_amdgcn_wmma_f32_16x16x32_bf16(false, a, false, b0,
                                                     (short)0, acc0, false, false);
      acc1 = __builtin_amdgcn_wmma_f32_16x16x32_bf16(false, a, false, b1,
                                                     (short)0, acc1, false, false);
    }
  }
  __syncthreads();
  {
    // C/D layout: VGPR r -> M = r + (lane<16 ? 0 : 8), N = lane%16
    int mb = mt * 16 + ((lane < 16) ? 0 : 8);
    int nc = lane & 15;
    #pragma unroll
    for (int r = 0; r < 8; ++r) {
      gates[g0 + 0][mb + r][nc] = acc0[r];
      gates[g0 + 1][mb + r][nc] = acc1[r];
    }
  }
  __syncthreads();
  // Peephole cell update for this block's 16 U-columns, all 64 batch rows.
  #pragma unroll
  for (int q = 0; q < 4; ++q) {
    int idx = q * NTHR + tid;     // 1024 items
    int m   = idx >> 4;
    int cl  = idx & 15;
    int u   = blk * 16 + cl;
    float gi = gates[0][m][cl] + bias[u];
    float gj = gates[1][m][cl] + bias[1024 + u];
    float gf = gates[2][m][cl] + bias[2048 + u];
    float go = gates[3][m][cl] + bias[3072 + u];
    float cv = c[m * 1024 + u];
    float cn = sigm(gf + 0.8f + cv * pf[u]) * cv + sigm(gi + cv * pi[u]) * tanh_f(gj);
    float hn = sigm(go + cn * po[u]) * tanh_f(cn);
    c[m * 1024 + u]  = cn;
    hw[m * 1024 + u] = hn;
  }
}

// Output projection: out = tanh(h1 @ W_lin + b_lin). Blocks 0..31 each own a
// 16-wide N tile (N=512). Waves 0..3 compute, all 8 waves help pack A.
__device__ void out_gemm(v16bf* Abuf, const float* h1n, const v16bf* Blp,
                         const float* b_lin, float* outp, float* pout, int blk)
{
  const int tid  = threadIdx.x;
  const int lane = tid & 31;
  const int wv   = tid >> 5;
  ASrc asrc;
  asrc.p0 = h1n; asrc.p1 = h1n; asrc.p2 = h1n;
  asrc.e0 = 1024; asrc.e1 = 2048;
  asrc.s0 = 1024; asrc.s1 = 1024; asrc.s2 = 1024;
  v8f acc = {};
  for (int ch = 0; ch < 4; ++ch) {          // K = 1024
    __syncthreads();
    packA(Abuf, asrc, ch * 256);
    __syncthreads();
    if (wv < 4) {
      #pragma unroll 1
      for (int ktl = 0; ktl < 8; ++ktl) {
        int kt = ch * 8 + ktl;
        v16bf a = Abuf[(wv * 8 + ktl) * 32 + lane];
        v16bf b = Blp[((size_t)(blk * 32 + kt)) * 32 + lane];
        acc = __builtin_amdgcn_wmma_f32_16x16x32_bf16(false, a, false, b,
                                                      (short)0, acc, false, false);
      }
    }
  }
  if (wv < 4) {
    int n  = blk * 16 + (lane & 15);
    int mb = wv * 16 + ((lane < 16) ? 0 : 8);
    #pragma unroll
    for (int r = 0; r < 8; ++r) {
      float v = tanh_f(acc[r] + b_lin[n]);
      outp[(mb + r) * 512 + n] = v;
      pout[(mb + r) * 512 + n] = v;
    }
  }
}

// Persistent whole-sequence kernel: 64 blocks, 3 grid barriers per timestep.
__global__ __launch_bounds__(NTHR, 1) void lstm_kernel(
    const float* __restrict__ V_seq, const float* __restrict__ B_init,
    const float* __restrict__ init_state,
    const float* __restrict__ b0, const float* __restrict__ pi0,
    const float* __restrict__ pf0, const float* __restrict__ po0,
    const float* __restrict__ b1, const float* __restrict__ pi1,
    const float* __restrict__ pf1, const float* __restrict__ po1,
    const float* __restrict__ b_lin,
    const bf16* __restrict__ W0p, const bf16* __restrict__ W1p,
    const bf16* __restrict__ Wlp,
    float* c0, float* h0, float* c1, float* h1, float* pout,
    float* out, unsigned* bar, const int* lenP)
{
  __shared__ v16bf Abuf[1024];          // 32 KB A-fragment staging
  __shared__ float gates[4][64][16];    // 16 KB i/j/f/o exchange
  const int blk = blockIdx.x;
  const int T = lenP[0] - 1;
  unsigned gen = 0;

  // Initialize state from initial_state (bs, L, 2, U) and B_init.
  for (int idx = blk * NTHR + threadIdx.x; idx < 64 * 1024; idx += NBLK * NTHR) {
    int m = idx >> 10, u = idx & 1023;
    c0[idx] = init_state[m * 4096 + u];
    h0[idx] = init_state[m * 4096 + 1024 + u];   // h0 ping buffer 0
    c1[idx] = init_state[m * 4096 + 2048 + u];
    h1[idx] = init_state[m * 4096 + 3072 + u];   // h1 ping buffer 0
  }
  for (int idx = blk * NTHR + threadIdx.x; idx < 64 * 512; idx += NBLK * NTHR)
    pout[idx] = B_init[idx];
  gen += NBLK; grid_barrier(bar, gen);

  for (int t = 0; t < T; ++t) {
    const float* h0r = h0 + (t & 1) * 65536;
    float*       h0w = h0 + ((t & 1) ^ 1) * 65536;
    const float* h1r = h1 + (t & 1) * 65536;
    float*       h1w = h1 + ((t & 1) ^ 1) * 65536;
    const float* vt  = V_seq + (size_t)t * 64 * 512;

    // Layer 0: x = [v_t | prev_out | h0_{t-1}]  (K = 2048)
    ASrc a0; a0.p0 = vt; a0.p1 = pout; a0.p2 = h0r;
    a0.e0 = 512; a0.e1 = 1024; a0.s0 = 512; a0.s1 = 512; a0.s2 = 1024;
    gemm_cell(Abuf, gates, a0, (const v16bf*)W0p, b0, pi0, pf0, po0, c0, h0w, blk);
    gen += NBLK; grid_barrier(bar, gen);

    // Layer 1: x = [h0_t | h1_{t-1}]  (K = 2048)
    ASrc a1; a1.p0 = h0w; a1.p1 = h1r; a1.p2 = h1r;
    a1.e0 = 1024; a1.e1 = 2048; a1.s0 = 1024; a1.s1 = 1024; a1.s2 = 1024;
    gemm_cell(Abuf, gates, a1, (const v16bf*)W1p, b1, pi1, pf1, po1, c1, h1w, blk);
    gen += NBLK; grid_barrier(bar, gen);

    // Output projection + tanh
    if (blk < 32)
      out_gemm(Abuf, h1w, (const v16bf*)Wlp, b_lin, out + (size_t)t * 32768, pout, blk);
    gen += NBLK; grid_barrier(bar, gen);
  }
}

// One-time weight pack: fp32 row-major [K][N] -> bf16 WMMA B-operand tiles.
// Fragment fid = (nt*(K/32) + kt)*32 + lane; within a fragment element e is
// (K = kt*32 + (lane<16?0:8) + (e<8 ? e : e+8), N = nt*16 + lane%16).
__global__ void pack_w_kernel(const float* __restrict__ W, bf16* __restrict__ dst,
                              int K, int N) {
  int fid  = blockIdx.x * blockDim.x + threadIdx.x;
  int ktN  = K >> 5;
  int lane = fid & 31;
  int kt   = (fid >> 5) % ktN;
  int nt   = fid / (32 * ktN);
  if (nt >= (N >> 4)) return;
  int Ng = nt * 16 + (lane & 15);
  int kh = (lane < 16) ? 0 : 8;
  v16bf frag;
  #pragma unroll
  for (int e = 0; e < 16; ++e)
    frag[e] = (bf16)W[(size_t)(kt * 32 + kh + (e < 8 ? e : e + 8)) * N + Ng];
  ((v16bf*)dst)[fid] = frag;
}

extern "C" void kernel_launch(void* const* d_in, const int* in_sizes, int n_in,
                              void* d_out, int out_size, void* d_ws, size_t ws_size,
                              hipStream_t stream) {
  const float* V_seq      = (const float*)d_in[0];
  const float* B_init     = (const float*)d_in[1];
  const float* init_state = (const float*)d_in[2];
  const float* W0    = (const float*)d_in[3];
  const float* b0    = (const float*)d_in[4];
  const float* pi0   = (const float*)d_in[5];
  const float* pf0   = (const float*)d_in[6];
  const float* po0   = (const float*)d_in[7];
  const float* W1    = (const float*)d_in[8];
  const float* b1    = (const float*)d_in[9];
  const float* pi1   = (const float*)d_in[10];
  const float* pf1   = (const float*)d_in[11];
  const float* po1   = (const float*)d_in[12];
  const float* W_lin = (const float*)d_in[13];
  const float* b_lin = (const float*)d_in[14];
  const int*   lenP  = (const int*)d_in[15];

  char* ws = (char*)d_ws;
  size_t off = 0;
  auto take = [&](size_t bytes) -> char* {
    char* p = ws + off;
    off += (bytes + 255) & ~(size_t)255;
    return p;
  };
  unsigned* bar = (unsigned*)take(256);
  float* c0   = (float*)take((size_t)64 * 1024 * 4);
  float* h0   = (float*)take((size_t)2 * 64 * 1024 * 4);   // ping-pong
  float* c1   = (float*)take((size_t)64 * 1024 * 4);
  float* h1   = (float*)take((size_t)2 * 64 * 1024 * 4);   // ping-pong
  float* pout = (float*)take((size_t)64 * 512 * 4);
  bf16* W0p = (bf16*)take((size_t)256 * 64 * 512 * 2);     // 16 MB
  bf16* W1p = (bf16*)take((size_t)256 * 64 * 512 * 2);     // 16 MB
  bf16* Wlp = (bf16*)take((size_t)32 * 32 * 512 * 2);      // 1 MB

  hipMemsetAsync(bar, 0, 256, stream);
  pack_w_kernel<<<(256 * 64 * 32) / NTHR, NTHR, 0, stream>>>(W0, W0p, 2048, 4096);
  pack_w_kernel<<<(256 * 64 * 32) / NTHR, NTHR, 0, stream>>>(W1, W1p, 2048, 4096);
  pack_w_kernel<<<(32 * 32 * 32) / NTHR, NTHR, 0, stream>>>(W_lin, Wlp, 1024, 512);
  lstm_kernel<<<NBLK, NTHR, 0, stream>>>(V_seq, B_init, init_state,
                                         b0, pi0, pf0, po0,
                                         b1, pi1, pf1, po1, b_lin,
                                         W0p, W1p, Wlp,
                                         c0, h0, c1, h1, pout,
                                         (float*)d_out, bar, lenP);
}
